// HANModel_6588479832224
// MI455X (gfx1250) — matrix operand order repaired and checked by reading, not compiled, not run on hardware
//
#include <hip/hip_runtime.h>

#define HEADS 4
#define HID 128
#define DPH 32
#define OUT_F 64
#define NEGS 0.2f
#define NA 100000
#define NP 200000
#define NEDGE 1000000

typedef __attribute__((ext_vector_type(16))) _Float16 v16h;
typedef __attribute__((ext_vector_type(8)))  float    v8f;

#define LDS_STRIDE 144   // halves per LDS row: 128 + 16 pad (288B, 16B aligned)
#define ROWS_PB 64       // rows of X staged per block

// ---------------------------------------------------------------------------
// WMMA fragment gathers (wave32); A and B both use contiguous k = 16*(lane/16) + s
// ---------------------------------------------------------------------------
__device__ __forceinline__ v16h load_A_frag(const _Float16* sA, int rowTile,
                                            int kOff, int lane) {
  const bool lo = lane < 16;
  const int  m  = lane & 15;
  const _Float16* rp = sA + (size_t)(rowTile * 16 + m) * LDS_STRIDE;
  const int b1 = kOff + (lo ? 0 : 16);   // VGPR0..3 -> K pairs
  const int b2 = kOff + (lo ? 8 : 24);   // VGPR4..7 -> K pairs
  v16h a;
#pragma unroll
  for (int p = 0; p < 4; ++p) {
    a[2 * p]     = rp[b1 + 2 * p];
    a[2 * p + 1] = rp[b1 + 2 * p + 1];
  }
#pragma unroll
  for (int p = 0; p < 4; ++p) {
    a[8 + 2 * p]     = rp[b2 + 2 * p];
    a[8 + 2 * p + 1] = rp[b2 + 2 * p + 1];
  }
  return a;
}

__device__ __forceinline__ v16h load_B_frag(const _Float16* sWt, int colTile,
                                            int kOff, int lane) {
  const bool lo = lane < 16;
  const int  n  = lane & 15;
  const _Float16* rp = sWt + (size_t)(colTile * 16 + n) * LDS_STRIDE
                           + kOff + (lo ? 0 : 16);
  v16h b;
#pragma unroll
  for (int h = 0; h < 16; ++h) b[h] = rp[h];
  return b;
}

// Stage X rows (f32->f16) and W transposed (sWt[n][k] = W[k][n]) into LDS.
template <int NCOLS>
__device__ __forceinline__ void stage_tiles(const float* __restrict__ X,
                                            const float* __restrict__ W,
                                            int row0, int N, int tid,
                                            _Float16* sA, _Float16* sWt) {
  constexpr int K = 128;
  for (int i = tid; i < ROWS_PB * (K / 4); i += 256) {
    const int r  = i / (K / 4);
    const int c4 = (i % (K / 4)) * 4;
    float4 v = make_float4(0.f, 0.f, 0.f, 0.f);
    if (row0 + r < N)
      v = ((const float4*)(X + (size_t)(row0 + r) * K))[c4 >> 2];
    _Float16* dp = sA + (size_t)r * LDS_STRIDE + c4;
    dp[0] = (_Float16)v.x; dp[1] = (_Float16)v.y;
    dp[2] = (_Float16)v.z; dp[3] = (_Float16)v.w;
  }
  for (int i = tid; i < K * (NCOLS / 4); i += 256) {
    const int k  = i / (NCOLS / 4);
    const int n4 = (i % (NCOLS / 4)) * 4;
    const float4 v = ((const float4*)(W + (size_t)k * NCOLS))[n4 >> 2];
    sWt[(size_t)(n4 + 0) * LDS_STRIDE + k] = (_Float16)v.x;
    sWt[(size_t)(n4 + 1) * LDS_STRIDE + k] = (_Float16)v.y;
    sWt[(size_t)(n4 + 2) * LDS_STRIDE + k] = (_Float16)v.z;
    sWt[(size_t)(n4 + 3) * LDS_STRIDE + k] = (_Float16)v.w;
  }
}

// ---------------------------------------------------------------------------
// Y[N,NCOLS] = X[N,128] @ W[128,NCOLS] + bias   (f16 WMMA, f32 accumulate)
// Each wave owns one 16-column strip: B fragments loaded once, reused
// across all row tiles.
// ---------------------------------------------------------------------------
template <int NCOLS>
__global__ void __launch_bounds__(256)
wmma_gemm_bias(const float* __restrict__ X, const float* __restrict__ W,
               const float* __restrict__ bias, float* __restrict__ Y, int N) {
  __shared__ _Float16 sA[ROWS_PB * LDS_STRIDE];
  __shared__ _Float16 sWt[NCOLS * LDS_STRIDE];
  const int tid  = threadIdx.x;
  const int row0 = blockIdx.x * ROWS_PB;
  stage_tiles<NCOLS>(X, W, row0, N, tid, sA, sWt);
  __syncthreads();

  const int wave = tid >> 5, lane = tid & 31;
  const bool lo = lane < 16;
  const int nn = lane & 15;
  constexpr int CT    = NCOLS / 16;   // column tiles (8 or 4)
  constexpr int RSTEP = 8 / CT;       // row-tile stride across waves sharing j

  const int j      = wave % CT;       // fixed column strip per wave
  const int rstart = wave / CT;

  // Hoisted B fragments for this column strip (4 K-tiles)
  v16h bf[4];
#pragma unroll
  for (int kt = 0; kt < 4; ++kt) bf[kt] = load_B_frag(sWt, j, kt * 32, lane);
  const float bcol = bias[j * 16 + nn];

  for (int r = rstart; r < ROWS_PB / 16; r += RSTEP) {
    v8f c;
#pragma unroll
    for (int g = 0; g < 8; ++g) c[g] = bcol;
#pragma unroll
    for (int kt = 0; kt < 4; ++kt) {
      v16h a = load_A_frag(sA, r, kt * 32, lane);
      c = __builtin_amdgcn_wmma_f32_16x16x32_f16(false, a, false, bf[kt],
                                                 (short)0, c, false, false);
    }
#pragma unroll
    for (int g = 0; g < 8; ++g) {
      const int m   = g + (lo ? 0 : 8);
      const int row = row0 + r * 16 + m;
      if (row < N) Y[(size_t)row * NCOLS + j * 16 + nn] = c[g];
    }
  }
}

// ---------------------------------------------------------------------------
// partial[col] += sum_rows tanh( (O @ Wk + bk)[row,col] )   (row-masked)
// ---------------------------------------------------------------------------
__global__ void __launch_bounds__(256)
wmma_score_tanh(const float* __restrict__ O, const float* __restrict__ Wk,
                const float* __restrict__ bk, float* __restrict__ partial,
                int N) {
  __shared__ _Float16 sA[ROWS_PB * LDS_STRIDE];
  __shared__ _Float16 sWt[128 * LDS_STRIDE];
  const int tid  = threadIdx.x;
  const int row0 = blockIdx.x * ROWS_PB;
  stage_tiles<128>(O, Wk, row0, N, tid, sA, sWt);
  __syncthreads();

  const int wave = tid >> 5, lane = tid & 31;
  const bool lo = lane < 16;
  const int nn = lane & 15;
  const int j = wave;                 // 8 waves, 8 column strips

  v16h bf[4];
#pragma unroll
  for (int kt = 0; kt < 4; ++kt) bf[kt] = load_B_frag(sWt, j, kt * 32, lane);
  const float bcol = bk[j * 16 + nn];

  float colsum = 0.f;                 // accumulated over all 4 row tiles
#pragma unroll
  for (int r = 0; r < 4; ++r) {
    v8f c;
#pragma unroll
    for (int g = 0; g < 8; ++g) c[g] = bcol;
#pragma unroll
    for (int kt = 0; kt < 4; ++kt) {
      v16h a = load_A_frag(sA, r, kt * 32, lane);
      c = __builtin_amdgcn_wmma_f32_16x16x32_f16(false, a, false, bf[kt],
                                                 (short)0, c, false, false);
    }
#pragma unroll
    for (int g = 0; g < 8; ++g) {
      const int row = row0 + r * 16 + g + (lo ? 0 : 8);
      if (row < N) colsum += tanhf(c[g]);
    }
  }
  atomicAdd(&partial[j * 16 + nn], colsum);
}

// ---------------------------------------------------------------------------
// Elementwise / edge kernels
// ---------------------------------------------------------------------------
__global__ void fill_k(float* __restrict__ p, float v, long long n) {
  for (long long i = (long long)blockIdx.x * blockDim.x + threadIdx.x; i < n;
       i += (long long)gridDim.x * blockDim.x)
    p[i] = v;
}

__global__ void relu_k(float* __restrict__ p, long long n) {
  for (long long i = (long long)blockIdx.x * blockDim.x + threadIdx.x; i < n;
       i += (long long)gridDim.x * blockDim.x)
    p[i] = fmaxf(p[i], 0.f);
}

// s_t[n*4+h] = dot(H[n, h*32 .. h*32+31], att_t[h*32 ..])
__global__ void node_scores(const float* __restrict__ H, int N,
                            const float* a0, float* s0,
                            const float* a1, float* s1,
                            const float* a2, float* s2,
                            const float* a3, float* s3) {
  const int i = blockIdx.x * blockDim.x + threadIdx.x;
  if (i >= N * HEADS) return;
  const int n = i >> 2, h = i & 3;
  const float* hp = H + (size_t)n * HID + h * DPH;
  float d0 = 0.f, d1 = 0.f, d2 = 0.f, d3 = 0.f;
#pragma unroll 8
  for (int k = 0; k < DPH; ++k) {
    const float x = hp[k];
    if (a0) d0 += x * a0[h * DPH + k];
    if (a1) d1 += x * a1[h * DPH + k];
    if (a2) d2 += x * a2[h * DPH + k];
    if (a3) d3 += x * a3[h * DPH + k];
  }
  if (s0) s0[i] = d0;
  if (s1) s1[i] = d1;
  if (s2) s2[i] = d2;
  if (s3) s3[i] = d3;
}

__device__ __forceinline__ float lrelu(float x) { return x > 0.f ? x : NEGS * x; }

__device__ __forceinline__ void atomicMaxF(float* a, float v) {
  if (v >= 0.f) atomicMax((int*)a, __float_as_int(v));
  else          atomicMin((unsigned int*)a, __float_as_uint(v));
}

__global__ void edge_max(const int* __restrict__ src, const int* __restrict__ dst,
                         const float* __restrict__ as, const float* __restrict__ ad,
                         float* __restrict__ m, int E) {
  const int e = blockIdx.x * blockDim.x + threadIdx.x;
  if (e >= E) return;
  const int s = src[e], d = dst[e];
#pragma unroll
  for (int h = 0; h < HEADS; ++h)
    atomicMaxF(&m[d * 4 + h], lrelu(as[s * 4 + h] + ad[d * 4 + h]));
}

__global__ void edge_den(const int* __restrict__ src, const int* __restrict__ dst,
                         const float* __restrict__ as, const float* __restrict__ ad,
                         const float* __restrict__ m, float* __restrict__ den, int E) {
  const int e = blockIdx.x * blockDim.x + threadIdx.x;
  if (e >= E) return;
  const int s = src[e], d = dst[e];
#pragma unroll
  for (int h = 0; h < HEADS; ++h) {
    const float v = lrelu(as[s * 4 + h] + ad[d * 4 + h]);
    atomicAdd(&den[d * 4 + h], __expf(v - m[d * 4 + h]));
  }
}

// wave per edge; lane owns 4 channels (float4 gather, 4x f32 atomic adds)
__global__ void __launch_bounds__(256)
edge_scatter(const int* __restrict__ src, const int* __restrict__ dst,
             const float* __restrict__ as, const float* __restrict__ ad,
             const float* __restrict__ m, const float* __restrict__ den,
             const float* __restrict__ Hsrc, float* __restrict__ out, int E) {
  const int wave = threadIdx.x >> 5, lane = threadIdx.x & 31;
  const int e = blockIdx.x * 8 + wave;
  if (e >= E) return;
  const int s = src[e], d = dst[e];
  const int h  = lane >> 3;           // channel 4*lane -> head (4*lane)/32
  const int ih = d * 4 + h;
  const float v = lrelu(as[s * 4 + h] + ad[ih]);
  const float w = __expf(v - m[ih]) / (den[ih] + 1e-16f);
  const float4 x = ((const float4*)(Hsrc + (size_t)s * HID))[lane];
  float* op = out + (size_t)d * HID + lane * 4;
  atomicAdd(op + 0, w * x.x);
  atomicAdd(op + 1, w * x.y);
  atomicAdd(op + 2, w * x.z);
  atomicAdd(op + 3, w * x.w);
}

// score_t = sum(q * partial_t / N); wts = softmax(score)
__global__ void score_finalize(const float* __restrict__ partial,
                               const float* __restrict__ q,
                               float* __restrict__ wts, float invN) {
  __shared__ float red0[128], red1[128];
  const int t = threadIdx.x;
  red0[t] = q[t] * partial[t] * invN;
  red1[t] = q[t] * partial[128 + t] * invN;
  __syncthreads();
  for (int s = 64; s > 0; s >>= 1) {
    if (t < s) { red0[t] += red0[t + s]; red1[t] += red1[t + s]; }
    __syncthreads();
  }
  if (t == 0) {
    const float s0 = red0[0], s1 = red1[0];
    const float mx = fmaxf(s0, s1);
    const float e0 = __expf(s0 - mx), e1 = __expf(s1 - mx);
    const float inv = 1.f / (e0 + e1);
    wts[0] = e0 * inv; wts[1] = e1 * inv;
  }
}

// a = wts[0]*a + wts[1]*b  (in-place combine of paper outputs)
__global__ void combine_scale(float* __restrict__ a, const float* __restrict__ b,
                              const float* __restrict__ wts, long long n) {
  const float w0 = wts[0], w1 = wts[1];
  for (long long i = (long long)blockIdx.x * blockDim.x + threadIdx.x; i < n;
       i += (long long)gridDim.x * blockDim.x)
    a[i] = w0 * a[i] + w1 * b[i];
}

// ---------------------------------------------------------------------------
extern "C" void kernel_launch(void* const* d_in, const int* in_sizes, int n_in,
                              void* d_out, int out_size, void* d_ws, size_t ws_size,
                              hipStream_t stream) {
  (void)in_sizes; (void)n_in; (void)out_size; (void)ws_size;
  const float* x_author = (const float*)d_in[0];
  const float* x_paper  = (const float*)d_in[1];
  const int* writes_src = (const int*)d_in[2];
  const int* writes_dst = (const int*)d_in[3];
  const int* wb_src     = (const int*)d_in[4];
  const int* wb_dst     = (const int*)d_in[5];
  const int* cites_src  = (const int*)d_in[6];
  const int* cites_dst  = (const int*)d_in[7];
  const float* W_pa = (const float*)d_in[8];
  const float* b_pa = (const float*)d_in[9];
  const float* W_pp = (const float*)d_in[10];
  const float* b_pp = (const float*)d_in[11];
  const float* att_s_w  = (const float*)d_in[12];
  const float* att_d_w  = (const float*)d_in[13];
  const float* att_s_wb = (const float*)d_in[14];
  const float* att_d_wb = (const float*)d_in[15];
  const float* att_s_c  = (const float*)d_in[16];
  const float* att_d_c  = (const float*)d_in[17];
  const float* W_k   = (const float*)d_in[18];
  const float* b_k   = (const float*)d_in[19];
  const float* q     = (const float*)d_in[20];
  const float* W_lin = (const float*)d_in[21];
  const float* b_lin = (const float*)d_in[22];
  float* out = (float*)d_out;

  char* ws = (char*)d_ws;
  size_t off = 0;
  auto alloc = [&](size_t bytes) -> float* {
    float* p = (float*)(ws + off);
    off += (bytes + 255) & ~(size_t)255;
    return p;
  };
  float* h_a    = alloc((size_t)NA * HID * 4);
  float* h_p    = alloc((size_t)NP * HID * 4);
  float* sa_ws  = alloc((size_t)NA * 4 * 4);   // author src scores (writes)
  float* sa_wbd = alloc((size_t)NA * 4 * 4);   // author dst scores (writtenby)
  float* sp_wd  = alloc((size_t)NP * 4 * 4);   // paper dst scores (writes)
  float* sp_wbs = alloc((size_t)NP * 4 * 4);   // paper src scores (writtenby)
  float* sp_cs  = alloc((size_t)NP * 4 * 4);   // paper src scores (cites)
  float* sp_cd  = alloc((size_t)NP * 4 * 4);   // paper dst scores (cites)
  float* m_buf  = alloc((size_t)NP * 4 * 4);   // reused per edge type
  float* den    = alloc((size_t)NP * 4 * 4);
  float* o_wr   = alloc((size_t)NP * HID * 4); // also holds combined papers
  float* o_wb   = alloc((size_t)NA * HID * 4);
  float* o_ci   = alloc((size_t)NP * HID * 4);
  float* part   = alloc(2 * 128 * 4);
  float* wts    = alloc(2 * 4);

  const int gA  = (NA + ROWS_PB - 1) / ROWS_PB;   // 1563
  const int gP  = (NP + ROWS_PB - 1) / ROWS_PB;   // 3125
  const int gE  = (NEDGE + 255) / 256;
  const int gEs = (NEDGE + 7) / 8;
  const float NINF = -__builtin_huge_valf();

  // 1) projections (WMMA)
  wmma_gemm_bias<HID><<<gA, 256, 0, stream>>>(x_author, W_pa, b_pa, h_a, NA);
  wmma_gemm_bias<HID><<<gP, 256, 0, stream>>>(x_paper, W_pp, b_pp, h_p, NP);

  // 2) per-node attention dots
  node_scores<<<(NA * 4 + 255) / 256, 256, 0, stream>>>(
      h_a, NA, att_s_w, sa_ws, att_d_wb, sa_wbd, nullptr, nullptr, nullptr, nullptr);
  node_scores<<<(NP * 4 + 255) / 256, 256, 0, stream>>>(
      h_p, NP, att_d_w, sp_wd, att_s_wb, sp_wbs, att_s_c, sp_cs, att_d_c, sp_cd);

  // 3) edge type "writes": author -> paper
  fill_k<<<1024, 256, 0, stream>>>(m_buf, NINF, (long long)NP * 4);
  fill_k<<<1024, 256, 0, stream>>>(den, 0.f, (long long)NP * 4);
  fill_k<<<4096, 256, 0, stream>>>(o_wr, 0.f, (long long)NP * HID);
  edge_max<<<gE, 256, 0, stream>>>(writes_src, writes_dst, sa_ws, sp_wd, m_buf, NEDGE);
  edge_den<<<gE, 256, 0, stream>>>(writes_src, writes_dst, sa_ws, sp_wd, m_buf, den, NEDGE);
  edge_scatter<<<gEs, 256, 0, stream>>>(writes_src, writes_dst, sa_ws, sp_wd,
                                        m_buf, den, h_a, o_wr, NEDGE);
  relu_k<<<4096, 256, 0, stream>>>(o_wr, (long long)NP * HID);

  // 4) edge type "writtenby": paper -> author
  fill_k<<<1024, 256, 0, stream>>>(m_buf, NINF, (long long)NA * 4);
  fill_k<<<1024, 256, 0, stream>>>(den, 0.f, (long long)NA * 4);
  fill_k<<<4096, 256, 0, stream>>>(o_wb, 0.f, (long long)NA * HID);
  edge_max<<<gE, 256, 0, stream>>>(wb_src, wb_dst, sp_wbs, sa_wbd, m_buf, NEDGE);
  edge_den<<<gE, 256, 0, stream>>>(wb_src, wb_dst, sp_wbs, sa_wbd, m_buf, den, NEDGE);
  edge_scatter<<<gEs, 256, 0, stream>>>(wb_src, wb_dst, sp_wbs, sa_wbd,
                                        m_buf, den, h_p, o_wb, NEDGE);
  relu_k<<<4096, 256, 0, stream>>>(o_wb, (long long)NA * HID);

  // 5) edge type "cites": paper -> paper
  fill_k<<<1024, 256, 0, stream>>>(m_buf, NINF, (long long)NP * 4);
  fill_k<<<1024, 256, 0, stream>>>(den, 0.f, (long long)NP * 4);
  fill_k<<<4096, 256, 0, stream>>>(o_ci, 0.f, (long long)NP * HID);
  edge_max<<<gE, 256, 0, stream>>>(cites_src, cites_dst, sp_cs, sp_cd, m_buf, NEDGE);
  edge_den<<<gE, 256, 0, stream>>>(cites_src, cites_dst, sp_cs, sp_cd, m_buf, den, NEDGE);
  edge_scatter<<<gEs, 256, 0, stream>>>(cites_src, cites_dst, sp_cs, sp_cd,
                                        m_buf, den, h_p, o_ci, NEDGE);
  relu_k<<<4096, 256, 0, stream>>>(o_ci, (long long)NP * HID);

  // 6) semantic attention over paper edge types (authors: T=1 -> identity)
  fill_k<<<1, 256, 0, stream>>>(part, 0.f, 256);
  wmma_score_tanh<<<gP, 256, 0, stream>>>(o_wr, W_k, b_k, part, NP);
  wmma_score_tanh<<<gP, 256, 0, stream>>>(o_ci, W_k, b_k, part + 128, NP);
  score_finalize<<<1, 128, 0, stream>>>(part, q, wts, 1.f / (float)NP);
  combine_scale<<<4096, 256, 0, stream>>>(o_wr, o_ci, wts, (long long)NP * HID);

  // 7) final shared linear, straight into d_out (authors first, then papers)
  wmma_gemm_bias<OUT_F><<<gA, 256, 0, stream>>>(o_wb, W_lin, b_lin, out, NA);
  wmma_gemm_bias<OUT_F><<<gP, 256, 0, stream>>>(o_wr, W_lin, b_lin,
                                                out + (size_t)NA * OUT_F, NP);
}
